// GRUEncoder_41712722379264
// MI455X (gfx1250) — matrix operand order repaired and checked by reading, not compiled, not run on hardware
//
#include <hip/hip_runtime.h>

#define BATCH 16384
#define INP   512
#define HID   1024
#define LDB   72   // padded LDS row stride (bf16 elems) for 64-wide K tiles

typedef __attribute__((ext_vector_type(16))) __bf16 v16bf;
typedef __attribute__((ext_vector_type(8)))  float  v8f;
typedef __attribute__((ext_vector_type(4)))  int    v4i;

#define AS1 __attribute__((address_space(1)))
#define AS3 __attribute__((address_space(3)))

struct alignas(8) bf16x4 { __bf16 v[4]; };

// ---------------- async global -> LDS (16B per lane), ASYNCcnt tracked ------

__device__ __forceinline__ void async_copy16(const void* g, void* l) {
#if defined(__has_builtin) && __has_builtin(__builtin_amdgcn_global_load_async_to_lds_b128)
  AS1 v4i* gp = (AS1 v4i*)(uintptr_t)g;
  AS3 v4i* lp = (AS3 v4i*)(uint32_t)(uintptr_t)l;  // LDS generic addr[31:0] = LDS offset
  __builtin_amdgcn_global_load_async_to_lds_b128(gp, lp, 0, 0);
#else
  uint32_t loff = (uint32_t)(uintptr_t)l;
  asm volatile("global_load_async_to_lds_b128 %0, %1, off"
               :: "v"(loff), "v"(g) : "memory");
#endif
}

__device__ __forceinline__ void wait_async0() {
#if defined(__has_builtin) && __has_builtin(__builtin_amdgcn_s_wait_asynccnt)
  __builtin_amdgcn_s_wait_asynccnt(0);
#else
  asm volatile("s_wait_asynccnt 0x0" ::: "memory");
#endif
}

// ---------------- prologue kernels ----------------

__global__ void f32_to_bf16_v4(const float4* __restrict__ src,
                               bf16x4* __restrict__ dst, int n4) {
  int i = blockIdx.x * blockDim.x + threadIdx.x;
  int stride = gridDim.x * blockDim.x;
  for (; i < n4; i += stride) {
    float4 f = src[i];
    bf16x4 o;
    o.v[0] = (__bf16)f.x; o.v[1] = (__bf16)f.y;
    o.v[2] = (__bf16)f.z; o.v[3] = (__bf16)f.w;
    dst[i] = o;
  }
}

// src: K x N (row-major, N contiguous) f32  ->  dst: N x K (K contiguous) bf16
__global__ __launch_bounds__(256) void transpose_to_bf16(
    const float* __restrict__ src, __bf16* __restrict__ dst, int K, int N) {
  __shared__ float tile[32][33];
  int n0 = blockIdx.x * 32, k0 = blockIdx.y * 32;
  int tx = threadIdx.x & 31, ty = threadIdx.x >> 5;  // 32 x 8
#pragma unroll
  for (int r = 0; r < 4; ++r)
    tile[ty + r * 8][tx] = src[(size_t)(k0 + ty + r * 8) * N + n0 + tx];
  __syncthreads();
#pragma unroll
  for (int r = 0; r < 4; ++r)
    dst[(size_t)(n0 + ty + r * 8) * K + k0 + tx] = (__bf16)tile[tx][ty + r * 8];
}

// ---------------- fragment loaders (LDS -> VGPR) ----------------

// A-frag: 16x32 bf16, M = lane%16, chunks at K = ko+half*8 and ko+16+half*8
__device__ __forceinline__ v16bf ldsA(const __bf16* Ax, int row, int half, int ko) {
  v16bf a;
  ((uint4*)&a)[0] = *(const uint4*)(Ax + row * LDB + ko + half * 8);
  ((uint4*)&a)[1] = *(const uint4*)(Ax + row * LDB + ko + 16 + half * 8);
  return a;
}
// B-frag: 32x16 bf16 (weights pre-transposed N x K): N = lane%16,
// chunks at K = ko+half*8 and ko+16+half*8
__device__ __forceinline__ v16bf ldsB(const __bf16* Bw, int row, int half, int ko) {
  v16bf b;
  ((uint4*)&b)[0] = *(const uint4*)(Bw + row * LDB + ko + half * 8);
  ((uint4*)&b)[1] = *(const uint4*)(Bw + row * LDB + ko + 16 + half * 8);
  return b;
}

__device__ __forceinline__ v8f wmma_bf16(v16bf a, v16bf b, v8f c) {
  return __builtin_amdgcn_wmma_f32_16x16x32_bf16(false, a, false, b,
                                                 (short)0, c, false, false);
}

__device__ __forceinline__ float sigmoidf(float x) {
  return 1.0f / (1.0f + __expf(-x));
}

// ---------------- kernel 1: r, z gates + x-part of candidate ----------------
// WG tile: 128 rows x 64 cols. 8 waves (4M x 2N), each wave 32x32, 3 gates.
// Double-buffered async staging of 64-wide K tiles.
__global__ __launch_bounds__(256) void gru_gates(
    const __bf16* __restrict__ xb, const __bf16* __restrict__ hb,
    const __bf16* __restrict__ Wxr, const __bf16* __restrict__ Wxz,
    const __bf16* __restrict__ Wxh, const __bf16* __restrict__ Whr,
    const __bf16* __restrict__ Whz, const float* __restrict__ bxr,
    const float* __restrict__ bxz, const float* __restrict__ bxh,
    const float* __restrict__ hidden, __bf16* __restrict__ rh,
    float* __restrict__ zf, float* __restrict__ tf) {
  __shared__ __align__(16) __bf16 Ax[2][128 * LDB];
  __shared__ __align__(16) __bf16 Bw[2][3][64 * LDB];

  const int tid = threadIdx.x;
  const int lane = tid & 31, wid = tid >> 5;
  const int wm = wid & 3, wn = wid >> 2;
  const int lm = lane & 15, half = lane >> 4;
  const int rowBase = blockIdx.y * 128;  // batch rows
  const int colBase = blockIdx.x * 64;   // hidden cols

  v8f acc[3][2][2];
#pragma unroll
  for (int g = 0; g < 3; ++g)
#pragma unroll
    for (int i = 0; i < 2; ++i)
#pragma unroll
      for (int j = 0; j < 2; ++j)
#pragma unroll
        for (int e = 0; e < 8; ++e) acc[g][i][j][e] = 0.0f;

  const __bf16* WB3[3] = {Wxr, Wxz, Wxh};
  const __bf16* WB2[2] = {Whr, Whz};

  // ---- phase 1: x @ {Wxr,Wxz,Wxh}, K = INP ----
  // prologue: stage 0 into buffer 0
#pragma unroll
  for (int c = tid; c < 128 * 8; c += 256) {
    int r = c >> 3, cc = c & 7;
    async_copy16(xb + (size_t)(rowBase + r) * INP + cc * 8,
                 Ax[0] + r * LDB + cc * 8);
  }
#pragma unroll
  for (int g = 0; g < 3; ++g)
#pragma unroll
    for (int c = tid; c < 64 * 8; c += 256) {
      int r = c >> 3, cc = c & 7;
      async_copy16(WB3[g] + (size_t)(colBase + r) * INP + cc * 8,
                   Bw[0][g] + r * LDB + cc * 8);
    }

  for (int s = 0; s < INP / 64; ++s) {
    const int cur = s & 1;
    wait_async0();
    __syncthreads();
    if (s + 1 < INP / 64) {
      const int k0 = (s + 1) * 64;
#pragma unroll
      for (int c = tid; c < 128 * 8; c += 256) {
        int r = c >> 3, cc = c & 7;
        async_copy16(xb + (size_t)(rowBase + r) * INP + k0 + cc * 8,
                     Ax[cur ^ 1] + r * LDB + cc * 8);
      }
#pragma unroll
      for (int g = 0; g < 3; ++g)
#pragma unroll
        for (int c = tid; c < 64 * 8; c += 256) {
          int r = c >> 3, cc = c & 7;
          async_copy16(WB3[g] + (size_t)(colBase + r) * INP + k0 + cc * 8,
                       Bw[cur ^ 1][g] + r * LDB + cc * 8);
        }
    }
#pragma unroll
    for (int sub = 0; sub < 2; ++sub) {
      const int ko = sub * 32;
      v16bf a0 = ldsA(Ax[cur], wm * 32 + lm, half, ko);
      v16bf a1 = ldsA(Ax[cur], wm * 32 + 16 + lm, half, ko);
#pragma unroll
      for (int g = 0; g < 3; ++g) {
        v16bf b0 = ldsB(Bw[cur][g], wn * 32 + lm, half, ko);
        v16bf b1 = ldsB(Bw[cur][g], wn * 32 + 16 + lm, half, ko);
        acc[g][0][0] = wmma_bf16(a0, b0, acc[g][0][0]);
        acc[g][0][1] = wmma_bf16(a0, b1, acc[g][0][1]);
        acc[g][1][0] = wmma_bf16(a1, b0, acc[g][1][0]);
        acc[g][1][1] = wmma_bf16(a1, b1, acc[g][1][1]);
      }
    }
  }

  // ---- phase 2: h @ {Whr,Whz}, K = HID ----
  // phase 1 had an even stage count, so buffer 0 is free to stage into.
#pragma unroll
  for (int c = tid; c < 128 * 8; c += 256) {
    int r = c >> 3, cc = c & 7;
    async_copy16(hb + (size_t)(rowBase + r) * HID + cc * 8,
                 Ax[0] + r * LDB + cc * 8);
  }
#pragma unroll
  for (int g = 0; g < 2; ++g)
#pragma unroll
    for (int c = tid; c < 64 * 8; c += 256) {
      int r = c >> 3, cc = c & 7;
      async_copy16(WB2[g] + (size_t)(colBase + r) * HID + cc * 8,
                   Bw[0][g] + r * LDB + cc * 8);
    }

  for (int s = 0; s < HID / 64; ++s) {
    const int cur = s & 1;
    wait_async0();
    __syncthreads();
    if (s + 1 < HID / 64) {
      const int k0 = (s + 1) * 64;
#pragma unroll
      for (int c = tid; c < 128 * 8; c += 256) {
        int r = c >> 3, cc = c & 7;
        async_copy16(hb + (size_t)(rowBase + r) * HID + k0 + cc * 8,
                     Ax[cur ^ 1] + r * LDB + cc * 8);
      }
#pragma unroll
      for (int g = 0; g < 2; ++g)
#pragma unroll
        for (int c = tid; c < 64 * 8; c += 256) {
          int r = c >> 3, cc = c & 7;
          async_copy16(WB2[g] + (size_t)(colBase + r) * HID + k0 + cc * 8,
                       Bw[cur ^ 1][g] + r * LDB + cc * 8);
        }
    }
#pragma unroll
    for (int sub = 0; sub < 2; ++sub) {
      const int ko = sub * 32;
      v16bf a0 = ldsA(Ax[cur], wm * 32 + lm, half, ko);
      v16bf a1 = ldsA(Ax[cur], wm * 32 + 16 + lm, half, ko);
#pragma unroll
      for (int g = 0; g < 2; ++g) {
        v16bf b0 = ldsB(Bw[cur][g], wn * 32 + lm, half, ko);
        v16bf b1 = ldsB(Bw[cur][g], wn * 32 + 16 + lm, half, ko);
        acc[g][0][0] = wmma_bf16(a0, b0, acc[g][0][0]);
        acc[g][0][1] = wmma_bf16(a0, b1, acc[g][0][1]);
        acc[g][1][0] = wmma_bf16(a1, b0, acc[g][1][0]);
        acc[g][1][1] = wmma_bf16(a1, b1, acc[g][1][1]);
      }
    }
  }

  // ---- epilogue: bias + sigmoid; emit r*h (bf16), z, t = x@Wxh+bxh ----
#pragma unroll
  for (int i = 0; i < 2; ++i)
#pragma unroll
    for (int j = 0; j < 2; ++j) {
      int col = colBase + wn * 32 + j * 16 + lm;
      float br = bxr[col], bz = bxz[col], bh = bxh[col];
#pragma unroll
      for (int v = 0; v < 8; ++v) {
        int row = rowBase + wm * 32 + i * 16 + half * 8 + v;
        size_t o = (size_t)row * HID + col;
        float hv = hidden[o];
        float rv = sigmoidf(acc[0][i][j][v] + br);
        float zv = sigmoidf(acc[1][i][j][v] + bz);
        rh[o] = (__bf16)(rv * hv);
        zf[o] = zv;
        tf[o] = acc[2][i][j][v] + bh;
      }
    }
}

// ---------------- kernel 2: (r*h) @ Whh, tanh, blend ----------------
__global__ __launch_bounds__(256) void gru_cand(
    const __bf16* __restrict__ rh, const __bf16* __restrict__ Whh,
    const float* __restrict__ tf, const float* __restrict__ zf,
    const float* __restrict__ hidden, float* __restrict__ out) {
  __shared__ __align__(16) __bf16 Ax[2][128 * LDB];
  __shared__ __align__(16) __bf16 Bw[2][64 * LDB];

  const int tid = threadIdx.x;
  const int lane = tid & 31, wid = tid >> 5;
  const int wm = wid & 3, wn = wid >> 2;
  const int lm = lane & 15, half = lane >> 4;
  const int rowBase = blockIdx.y * 128;
  const int colBase = blockIdx.x * 64;

  v8f acc[2][2];
#pragma unroll
  for (int i = 0; i < 2; ++i)
#pragma unroll
    for (int j = 0; j < 2; ++j)
#pragma unroll
      for (int e = 0; e < 8; ++e) acc[i][j][e] = 0.0f;

  // prologue stage
#pragma unroll
  for (int c = tid; c < 128 * 8; c += 256) {
    int r = c >> 3, cc = c & 7;
    async_copy16(rh + (size_t)(rowBase + r) * HID + cc * 8,
                 Ax[0] + r * LDB + cc * 8);
  }
#pragma unroll
  for (int c = tid; c < 64 * 8; c += 256) {
    int r = c >> 3, cc = c & 7;
    async_copy16(Whh + (size_t)(colBase + r) * HID + cc * 8,
                 Bw[0] + r * LDB + cc * 8);
  }

  for (int s = 0; s < HID / 64; ++s) {
    const int cur = s & 1;
    wait_async0();
    __syncthreads();
    if (s + 1 < HID / 64) {
      const int k0 = (s + 1) * 64;
#pragma unroll
      for (int c = tid; c < 128 * 8; c += 256) {
        int r = c >> 3, cc = c & 7;
        async_copy16(rh + (size_t)(rowBase + r) * HID + k0 + cc * 8,
                     Ax[cur ^ 1] + r * LDB + cc * 8);
      }
#pragma unroll
      for (int c = tid; c < 64 * 8; c += 256) {
        int r = c >> 3, cc = c & 7;
        async_copy16(Whh + (size_t)(colBase + r) * HID + k0 + cc * 8,
                     Bw[cur ^ 1] + r * LDB + cc * 8);
      }
    }
#pragma unroll
    for (int sub = 0; sub < 2; ++sub) {
      const int ko = sub * 32;
      v16bf a0 = ldsA(Ax[cur], wm * 32 + lm, half, ko);
      v16bf a1 = ldsA(Ax[cur], wm * 32 + 16 + lm, half, ko);
      v16bf b0 = ldsB(Bw[cur], wn * 32 + lm, half, ko);
      v16bf b1 = ldsB(Bw[cur], wn * 32 + 16 + lm, half, ko);
      acc[0][0] = wmma_bf16(a0, b0, acc[0][0]);
      acc[0][1] = wmma_bf16(a0, b1, acc[0][1]);
      acc[1][0] = wmma_bf16(a1, b0, acc[1][0]);
      acc[1][1] = wmma_bf16(a1, b1, acc[1][1]);
    }
  }

#pragma unroll
  for (int i = 0; i < 2; ++i)
#pragma unroll
    for (int j = 0; j < 2; ++j) {
      int col = colBase + wn * 32 + j * 16 + lm;
#pragma unroll
      for (int v = 0; v < 8; ++v) {
        int row = rowBase + wm * 32 + i * 16 + half * 8 + v;
        size_t o = (size_t)row * HID + col;
        float hc = tanhf(acc[i][j][v] + tf[o]);
        float zv = zf[o];
        float hv = hidden[o];
        out[o] = zv * hv + (1.0f - zv) * hc;
      }
    }
}

// ---------------- launcher ----------------

extern "C" void kernel_launch(void* const* d_in, const int* in_sizes, int n_in,
                              void* d_out, int out_size, void* d_ws,
                              size_t ws_size, hipStream_t stream) {
  const float* x      = (const float*)d_in[0];
  const float* hidden = (const float*)d_in[1];
  const float* Wxr    = (const float*)d_in[2];
  const float* bxr    = (const float*)d_in[3];
  const float* Whr    = (const float*)d_in[4];
  const float* Wxz    = (const float*)d_in[5];
  const float* bxz    = (const float*)d_in[6];
  const float* Whz    = (const float*)d_in[7];
  const float* Wxh    = (const float*)d_in[8];
  const float* bxh    = (const float*)d_in[9];
  const float* Whh    = (const float*)d_in[10];
  float* out = (float*)d_out;

  char* ws = (char*)d_ws;
  size_t oXB  = 0;                              // x  bf16 : B*I*2
  size_t oHB  = oXB  + (size_t)BATCH * INP * 2;
  size_t oWXR = oHB  + (size_t)BATCH * HID * 2;
  size_t oWXZ = oWXR + (size_t)INP * HID * 2;
  size_t oWXH = oWXZ + (size_t)INP * HID * 2;
  size_t oWHR = oWXH + (size_t)INP * HID * 2;
  size_t oWHZ = oWHR + (size_t)HID * HID * 2;
  size_t oWHH = oWHZ + (size_t)HID * HID * 2;
  size_t oRH  = oWHH + (size_t)HID * HID * 2;   // r*h bf16
  size_t oZ   = oRH  + (size_t)BATCH * HID * 2; // z   f32
  size_t oT   = oZ   + (size_t)BATCH * HID * 4; // x@Wxh+bxh f32

  __bf16* xb    = (__bf16*)(ws + oXB);
  __bf16* hb    = (__bf16*)(ws + oHB);
  __bf16* WxrT  = (__bf16*)(ws + oWXR);
  __bf16* WxzT  = (__bf16*)(ws + oWXZ);
  __bf16* WxhT  = (__bf16*)(ws + oWXH);
  __bf16* WhrT  = (__bf16*)(ws + oWHR);
  __bf16* WhzT  = (__bf16*)(ws + oWHZ);
  __bf16* WhhT  = (__bf16*)(ws + oWHH);
  __bf16* rhBuf = (__bf16*)(ws + oRH);
  float*  zBuf  = (float*)(ws + oZ);
  float*  tBuf  = (float*)(ws + oT);

  // convert activations to bf16
  int n4x = BATCH * INP / 4, n4h = BATCH * HID / 4;
  f32_to_bf16_v4<<<4096, 256, 0, stream>>>((const float4*)x, (bf16x4*)xb, n4x);
  f32_to_bf16_v4<<<4096, 256, 0, stream>>>((const float4*)hidden, (bf16x4*)hb, n4h);

  // transpose + convert weights: (K,N) f32 -> (N,K) bf16
  dim3 tgx(HID / 32, INP / 32);   // K=INP, N=HID
  dim3 tgh(HID / 32, HID / 32);   // K=HID, N=HID
  transpose_to_bf16<<<tgx, 256, 0, stream>>>(Wxr, WxrT, INP, HID);
  transpose_to_bf16<<<tgx, 256, 0, stream>>>(Wxz, WxzT, INP, HID);
  transpose_to_bf16<<<tgx, 256, 0, stream>>>(Wxh, WxhT, INP, HID);
  transpose_to_bf16<<<tgh, 256, 0, stream>>>(Whr, WhrT, HID, HID);
  transpose_to_bf16<<<tgh, 256, 0, stream>>>(Whz, WhzT, HID, HID);
  transpose_to_bf16<<<tgh, 256, 0, stream>>>(Whh, WhhT, HID, HID);

  dim3 grid(HID / 64, BATCH / 128);
  gru_gates<<<grid, 256, 0, stream>>>(xb, hb, WxrT, WxzT, WxhT, WhrT, WhzT,
                                      bxr, bxz, bxh, hidden, rhBuf, zBuf, tBuf);
  gru_cand<<<grid, 256, 0, stream>>>(rhBuf, WhhT, tBuf, zBuf, hidden, out);
}
